// HAttention1D_37804302139518
// MI455X (gfx1250) — compile-verified, matrix operand order checked by reading, not run on hardware
//
#include <hip/hip_runtime.h>
#include <math.h>

// ---------- types ----------
typedef float  v8f  __attribute__((ext_vector_type(8)));
typedef float  v2f  __attribute__((ext_vector_type(2)));
typedef __bf16 bf16_t;
typedef bf16_t v16bf __attribute__((ext_vector_type(16)));
typedef bf16_t v8bf  __attribute__((ext_vector_type(8)));
typedef bf16_t v2bf  __attribute__((ext_vector_type(2)));

#define HEADS 16
#define DH    64
#define BSZ   16
#define SEQ   8192
#define BATCH 4
#define DIM   1024
#define INNER 1024
#define BH    (BATCH*HEADS)
#define NLEV  9

// xor-butterfly lane exchange via ds_swizzle (group-of-32 mode):
// pattern = xor_mask<<10 | or_mask<<5 | and_mask ; and_mask=0x1f keeps all lanes
template <int XORMASK>
static __device__ inline float swz_xor(float x) {
  return __int_as_float(
      __builtin_amdgcn_ds_swizzle(__float_as_int(x), (XORMASK << 10) | 0x1f));
}

// load a 16x32-bf16 WMMA A/B fragment (row K-contiguous in LDS, row at rowptr)
// lane layout: elems 0..7 -> K = half*8 + 0..7 ; elems 8..15 -> K = 16 + half*8 + 0..7
static __device__ inline v16bf load_frag16(const bf16_t* rowptr, int hf) {
  v16bf f;
  const v8bf* p0 = (const v8bf*)(rowptr + hf * 8);
  const v8bf* p1 = (const v8bf*)(rowptr + 16 + hf * 8);
#pragma unroll
  for (int i = 0; i < 8; ++i) { f[i] = (*p0)[i]; f[i + 8] = (*p1)[i]; }
  return f;
}

// one 128x128x32 compute step: 8 WMMAs per wave (4 m-tiles x 2 n-tiles)
static __device__ inline void gemm_step(const bf16_t (*__restrict__ As)[40],
                                        const bf16_t (*__restrict__ Bs)[40],
                                        v8f acc[4][2], int wm, int wn, int fr, int hf)
{
  v16bf bfrag[2];
#pragma unroll
  for (int nt = 0; nt < 2; ++nt)
    bfrag[nt] = load_frag16(&Bs[wn + nt * 16 + fr][0], hf);
#pragma unroll
  for (int mt = 0; mt < 4; ++mt) {
    v16bf afrag = load_frag16(&As[wm + mt * 16 + fr][0], hf);
#pragma unroll
    for (int nt = 0; nt < 2; ++nt)
      acc[mt][nt] = __builtin_amdgcn_wmma_f32_16x16x32_bf16(
          false, afrag, false, bfrag[nt], (short)0, acc[mt][nt], false, false);
  }
}

// ---------- GEMM: C[M,N] = A[M,K] @ B[K,N] (+bias), f32 in/out, bf16 WMMA ----------
// 128x128 tile / WG (8 waves, each 64x32), BK=32, double-buffered LDS,
// float4 global loads, packed-vector bf16 LDS stores (b128/b32, not b16).
template <bool HAS_BIAS>
__global__ __launch_bounds__(256)
void gemm_bf16_128x128(const float* __restrict__ A, const float* __restrict__ B,
                       const float* __restrict__ bias, float* __restrict__ C,
                       int M, int N, int K)
{
  __shared__ __align__(16) bf16_t As[2][128][40];  // [stage][m][k], +8 pad
  __shared__ __align__(16) bf16_t Bs[2][128][40];  // [stage][n][k] (B transposed)
  const int tid  = threadIdx.x;
  const int wave = tid >> 5, lane = tid & 31;
  const int bm = blockIdx.y * 128, bn = blockIdx.x * 128;
  const int wm = (wave >> 2) * 64;   // 2 waves along M (64 rows each)
  const int wn = (wave & 3) * 32;    // 4 waves along N (32 cols each)
  const int fr = lane & 15, hf = lane >> 4;

  // per-thread tile-load mappings
  const int a_row = tid >> 1, a_col = (tid & 1) * 16;         // A: 16 contiguous floats
  const int b_kk  = (tid >> 4) * 2, b_n0 = (tid & 15) * 8;    // B: 2 rows x 8 floats

  const v8f zero8 = {0.f,0.f,0.f,0.f,0.f,0.f,0.f,0.f};
  v8f acc[4][2];
#pragma unroll
  for (int i = 0; i < 4; ++i)
#pragma unroll
    for (int j = 0; j < 2; ++j) acc[i][j] = zero8;

  float4 ra[4];           // A prefetch regs: 16 floats
  float4 rb0[2], rb1[2];  // B prefetch regs: rows kk, kk+1, 8 floats each

  auto gload = [&](int k0) {
    const float* sa = A + (size_t)(bm + a_row) * K + k0 + a_col;
#pragma unroll
    for (int i = 0; i < 4; ++i) ra[i] = ((const float4*)sa)[i];
    const float* sb0 = B + (size_t)(k0 + b_kk) * N + bn + b_n0;
    const float* sb1 = sb0 + N;
#pragma unroll
    for (int i = 0; i < 2; ++i) { rb0[i] = ((const float4*)sb0)[i]; rb1[i] = ((const float4*)sb1)[i]; }
  };
  auto lstore = [&](int st) {
    // A: two v8bf (16B) packed-convert stores
    const float* fa = (const float*)ra;
    v8bf pa0, pa1;
#pragma unroll
    for (int i = 0; i < 8; ++i) { pa0[i] = (bf16_t)fa[i]; pa1[i] = (bf16_t)fa[8 + i]; }
    *(v8bf*)&As[st][a_row][a_col]     = pa0;
    *(v8bf*)&As[st][a_row][a_col + 8] = pa1;
    // B transposed: 8 v2bf (4B) stores, pairing k and k+1
    const float* f0 = (const float*)rb0;
    const float* f1 = (const float*)rb1;
#pragma unroll
    for (int j = 0; j < 8; ++j) {
      v2bf p; p[0] = (bf16_t)f0[j]; p[1] = (bf16_t)f1[j];
      *(v2bf*)&Bs[st][b_n0 + j][b_kk] = p;
    }
  };

  const int nkt = K >> 5;
  gload(0);
  lstore(0);
  __syncthreads();
  for (int kt = 0; kt < nkt - 1; ++kt) {
    const int cur = kt & 1, nxt = cur ^ 1;
    gload((kt + 1) << 5);                 // prefetch next tile (overlaps WMMAs)
    gemm_step(As[cur], Bs[cur], acc, wm, wn, fr, hf);
    lstore(nxt);                          // nxt buffer unused by any wave this iter
    __syncthreads();                      // one barrier per iteration
  }
  gemm_step(As[(nkt - 1) & 1], Bs[(nkt - 1) & 1], acc, wm, wn, fr, hf);

  // epilogue: C/D layout -> row = r + 8*half, col = lane&15 (within 16x16 tile)
  float bv[2] = {0.f, 0.f};
  if constexpr (HAS_BIAS) {
#pragma unroll
    for (int nt = 0; nt < 2; ++nt) bv[nt] = bias[bn + wn + nt * 16 + fr];
  }
#pragma unroll
  for (int mt = 0; mt < 4; ++mt)
#pragma unroll
    for (int nt = 0; nt < 2; ++nt)
#pragma unroll
      for (int r = 0; r < 8; ++r) {
        int row = bm + wm + mt * 16 + r + 8 * hf;
        int col = bn + wn + nt * 16 + fr;
        C[(size_t)row * N + col] = acc[mt][nt][r] + bv[nt];
      }
}

// ---------- pyramid downsample: level 0 (qkv layout) -> level 1 (packed) ----------
__global__ __launch_bounds__(256)
void downsample_qkv(const float* __restrict__ qkv, float* __restrict__ q1,
                    float* __restrict__ k1, float* __restrict__ v1, int n_out)
{
  size_t idx = (size_t)blockIdx.x * 256 + threadIdx.x;
  int d = (int)(idx & 63);
  size_t t = idx >> 6;
  int j  = (int)(t % n_out);
  int bh = (int)(t / n_out);
  size_t b = (size_t)(bh >> 4), h = (size_t)(bh & 15);
  size_t base = (b * SEQ + (size_t)2 * j) * (3 * INNER) + h * 64 + d;
  float qa = qkv[base],             qb = qkv[base + 3*INNER];
  float ka = qkv[base + INNER],     kb = qkv[base + INNER + 3*INNER];
  float va = qkv[base + 2*INNER],   vb = qkv[base + 2*INNER + 3*INNER];
  q1[idx] = 0.5f * (qa + qb);
  k1[idx] = 0.5f * (ka + kb);
  v1[idx] = va + vb;
}

// ---------- pyramid downsample: packed (BH, 2n, 64) -> packed (BH, n, 64) ----------
__global__ __launch_bounds__(256)
void downsample_pyr(const float* __restrict__ qi, const float* __restrict__ ki,
                    const float* __restrict__ vi, float* __restrict__ qo,
                    float* __restrict__ ko, float* __restrict__ vo, int n_out)
{
  size_t idx = (size_t)blockIdx.x * 256 + threadIdx.x;
  int d = (int)(idx & 63);
  size_t t = idx >> 6;
  int j  = (int)(t % n_out);
  int bh = (int)(t / n_out);
  size_t base = ((size_t)bh * (2 * n_out) + (size_t)2 * j) * 64 + d;
  qo[idx] = 0.5f * (qi[base] + qi[base + 64]);
  ko[idx] = 0.5f * (ki[base] + ki[base + 64]);
  vo[idx] = vi[base] + vi[base + 64];
}

// ---------- per-level block attention: one wave per 16-token block ----------
// S = q k^T (f32 WMMA 16x16x4, K=64), rowmax/exp/rowsum via ds_swizzle butterflies,
// y = A v (f32 WMMA 16x16x4, K=16), Yl/Al written unnormalized.
__global__ __launch_bounds__(128)
void attn_level(const float* __restrict__ qsrc, const float* __restrict__ ksrc,
                const float* __restrict__ vsrc, int n_l, int level0, int flip,
                float scale, float* __restrict__ Yl, float* __restrict__ Al)
{
  __shared__ __align__(16) float lds[4][3 * 16 * 68 + 16 * 17];
  const int wave = threadIdx.x >> 5, lane = threadIdx.x & 31;
  const int nblk = n_l / BSZ;
  const int gblk = blockIdx.x * 4 + wave;
  const int bh = gblk / nblk, blk = gblk % nblk;
  const int kblk = flip ? (blk ^ 1) : blk;  // _flip_every_two on k/v blocks

  float* qs  = &lds[wave][0];
  float* ks  = qs + 16 * 68;   // row stride 68 floats: conflict-free frag reads
  float* vs  = ks + 16 * 68;
  float* as_ = vs + 16 * 68;   // exp'd 16x16 attention tile, stride 17

  // cooperative 16x64 tile loads (float4-vectorized), one wave per block
  const int lrow = lane >> 1, lcol = (lane & 1) * 32;
  {
    size_t qbase, kvbase;
    if (level0) {
      size_t b = (size_t)(bh >> 4), h = (size_t)(bh & 15);
      qbase  = (b * SEQ + (size_t)(blk  * 16 + lrow)) * (3 * INNER) + h * 64 + lcol;
      kvbase = (b * SEQ + (size_t)(kblk * 16 + lrow)) * (3 * INNER) + h * 64 + lcol;
    } else {
      qbase  = ((size_t)bh * n_l + (size_t)(blk  * 16 + lrow)) * 64 + lcol;
      kvbase = ((size_t)bh * n_l + (size_t)(kblk * 16 + lrow)) * 64 + lcol;
    }
    const float4* q4 = (const float4*)(qsrc + qbase);
    const float4* k4 = (const float4*)(ksrc + kvbase);
    const float4* v4 = (const float4*)(vsrc + kvbase);
    float4* qd = (float4*)(qs + lrow * 68 + lcol);
    float4* kd = (float4*)(ks + lrow * 68 + lcol);
    float4* vd = (float4*)(vs + lrow * 68 + lcol);
#pragma unroll
    for (int c = 0; c < 8; ++c) { qd[c] = q4[c]; kd[c] = k4[c]; vd[c] = v4[c]; }
  }
  // wave-private LDS region; same-wave DS ops are in order -> no barrier needed

  const int m = lane & 15, hf = lane >> 4;
  const v8f zero8 = {0.f,0.f,0.f,0.f,0.f,0.f,0.f,0.f};

  // S = q k^T : A-frag elem e -> K = 2*half + e ; B column n = lane&15, B[k][n] = kb[n][k]
  v8f sacc = zero8;
#pragma unroll
  for (int c = 0; c < 16; ++c) {
    const int k0 = c * 4 + hf * 2;
    v2f a, b;
    a.x = qs[m * 68 + k0]; a.y = qs[m * 68 + k0 + 1];
    b.x = ks[m * 68 + k0]; b.y = ks[m * 68 + k0 + 1];
    sacc = __builtin_amdgcn_wmma_f32_16x16x4_f32(
        false, a, false, b, (short)0, sacc, false, false);
  }

  // per-row (elem r -> row r+8*half, col = lane&15) max / exp / sum
  // xor butterflies with masks 1,2,4,8 stay inside each 16-lane half of the wave
#pragma unroll
  for (int r = 0; r < 8; ++r) {
    float sv = sacc[r];
    float mx = sv;
    mx = fmaxf(mx, swz_xor<1>(mx));
    mx = fmaxf(mx, swz_xor<2>(mx));
    mx = fmaxf(mx, swz_xor<4>(mx));
    mx = fmaxf(mx, swz_xor<8>(mx));
    float e = __expf((sv - mx) * scale);  // q was pre-scaled in ref; fold linearly here
    float sm = e;
    sm += swz_xor<1>(sm);
    sm += swz_xor<2>(sm);
    sm += swz_xor<4>(sm);
    sm += swz_xor<8>(sm);
    as_[(r + 8 * hf) * 17 + m] = e;
    if (m == 0) Al[(size_t)bh * n_l + (size_t)blk * 16 + r + 8 * hf] = sm;
  }

  // y = A v : 4 N-tiles of 16, K=16 in chunks of 4
#pragma unroll
  for (int nt = 0; nt < 4; ++nt) {
    v8f y = zero8;
#pragma unroll
    for (int c = 0; c < 4; ++c) {
      const int k0 = c * 4 + hf * 2;
      v2f a, b;
      a.x = as_[m * 17 + k0];     a.y = as_[m * 17 + k0 + 1];
      b.x = vs[k0 * 68 + nt * 16 + m];
      b.y = vs[(k0 + 1) * 68 + nt * 16 + m];
      y = __builtin_amdgcn_wmma_f32_16x16x4_f32(
          false, a, false, b, (short)0, y, false, false);
    }
#pragma unroll
    for (int r = 0; r < 8; ++r)
      Yl[((size_t)bh * n_l + (size_t)blk * 16 + r + 8 * hf) * 64 + nt * 16 + m] = y[r];
  }
}

// ---------- combine levels (repeat-upsample == gather i>>l), normalize, head-merge ----------
struct CombineArgs { const float* Y[NLEV]; const float* A[NLEV]; };

__global__ __launch_bounds__(256)
void combine_kernel(CombineArgs args, float* __restrict__ ctx)
{
  size_t idx = (size_t)blockIdx.x * 256 + threadIdx.x;  // over BH*SEQ*DH
  int d = (int)(idx & 63);
  size_t t = idx >> 6;
  int i  = (int)(t % SEQ);
  int bh = (int)(t / SEQ);
  float ys = 0.f, as = 0.f;
#pragma unroll
  for (int l = 0; l < NLEV; ++l) {
    int nl = SEQ >> l;
    int il = i >> l;
    ys += args.Y[l][((size_t)bh * nl + il) * 64 + d];
    as += args.A[l][(size_t)bh * nl + il];
  }
  float val = ys / (as + 1e-8f);
  // (b, h, i, d) -> ctx[b][i][h*64+d]
  ctx[(((size_t)(bh >> 4)) * SEQ + i) * (size_t)INNER + (size_t)(bh & 15) * 64 + d] = val;
}

// ---------- host ----------
extern "C" void kernel_launch(void* const* d_in, const int* in_sizes, int n_in,
                              void* d_out, int out_size, void* d_ws, size_t ws_size,
                              hipStream_t stream)
{
  const float* x     = (const float*)d_in[0];
  const float* w_qkv = (const float*)d_in[1];
  const float* w_out = (const float*)d_in[2];
  const float* b_out = (const float*)d_in[3];
  float* out = (float*)d_out;
  float* ws  = (float*)d_ws;

  size_t off = 0;
  float* qkv = ws; off += (size_t)BATCH * SEQ * 3 * INNER;       // (b, n, 3*inner)
  float* pq[NLEV]; float* pk[NLEV]; float* pv[NLEV];
  pq[0] = pk[0] = pv[0] = nullptr;                               // level 0 lives in qkv
  for (int l = 1; l < NLEV; ++l) {
    size_t sz = (size_t)BH * (SEQ >> l) * DH;
    pq[l] = ws + off; off += sz;
    pk[l] = ws + off; off += sz;
    pv[l] = ws + off; off += sz;
  }
  CombineArgs ca;
  float* Y[NLEV]; float* Aa[NLEV];
  for (int l = 0; l < NLEV; ++l) {
    size_t nl = (size_t)(SEQ >> l);
    Y[l]  = ws + off; off += (size_t)BH * nl * DH;
    Aa[l] = ws + off; off += (size_t)BH * nl;
    ca.Y[l] = Y[l]; ca.A[l] = Aa[l];
  }
  float* ctx = qkv;  // safe reuse: combine runs after all reads of qkv

  // 1) qkv = x @ w_qkv            (M=32768, N=3072, K=1024)
  gemm_bf16_128x128<false><<<dim3(3 * INNER / 128, BATCH * SEQ / 128), 256, 0, stream>>>(
      x, w_qkv, nullptr, qkv, BATCH * SEQ, 3 * INNER, DIM);

  // 2) build pyramid (q,k mean; v sum)
  {
    int n_out = SEQ >> 1;
    size_t cnt = (size_t)BH * n_out * DH;
    downsample_qkv<<<(unsigned)(cnt / 256), 256, 0, stream>>>(qkv, pq[1], pk[1], pv[1], n_out);
    for (int l = 2; l < NLEV; ++l) {
      n_out = SEQ >> l;
      cnt = (size_t)BH * n_out * DH;
      downsample_pyr<<<(unsigned)(cnt / 256), 256, 0, stream>>>(
          pq[l - 1], pk[l - 1], pv[l - 1], pq[l], pk[l], pv[l], n_out);
    }
  }

  // 3) block attention per level (flip k/v pairs on all coarse levels)
  const float scale = 0.125f;  // DIM_HEAD^-0.5
  for (int l = 0; l < NLEV; ++l) {
    int n_l = SEQ >> l;
    int total_blocks = BH * (n_l / BSZ);
    if (l == 0)
      attn_level<<<total_blocks / 4, 128, 0, stream>>>(
          qkv, qkv + INNER, qkv + 2 * INNER, n_l, 1, 0, scale, Y[0], Aa[0]);
    else
      attn_level<<<total_blocks / 4, 128, 0, stream>>>(
          pq[l], pk[l], pv[l], n_l, 0, 1, scale, Y[l], Aa[l]);
  }

  // 4) combine + normalize + head-merge into ctx (b, n, inner)
  combine_kernel<<<(unsigned)(((size_t)BH * SEQ * DH) / 256), 256, 0, stream>>>(ca, ctx);

  // 5) out = ctx @ w_out + b_out  (M=32768, N=1024, K=1024)
  gemm_bf16_128x128<true><<<dim3(DIM / 128, BATCH * SEQ / 128), 256, 0, stream>>>(
      ctx, w_out, b_out, out, BATCH * SEQ, DIM, INNER);
}
